// Attention_37314675868038
// MI455X (gfx1250) — compile-verified, matrix operand order checked by reading, not run on hardware
//
#include <hip/hip_runtime.h>
#include <hip/hip_bf16.h>
#include <math.h>

// ---------------------------------------------------------------------------
// Bahdanau attention, MI455X (gfx1250, wave32, WMMA)
//   B=64, N=2048, ENC=512, DEC=256
//   out = [context (64x512) | weights (64x2048)] fp32
// Pipeline:
//   K1: h_proj = hidden @ Wh.T + attn_b            (tiny, fp32 SIMT)  -> ws
//   K2: Webf   = bf16(attn_w[:,256:])              (256KB, L2-hot)    -> ws
//   K3: scores via V_WMMA_F32_16X16X32_BF16, fused tanh + v_w dot
//       -> written into the weights region of d_out
//   K4: softmax in-place over n
//   K5: context = weights^T @ enc_out (coalesced fp32 stream)
// ---------------------------------------------------------------------------

typedef __attribute__((ext_vector_type(16))) __bf16 v16bf;
typedef __attribute__((ext_vector_type(8)))  float  v8f;

#define B_DIM   64
#define N_DIM   2048
#define ENC_DIM 512
#define DEC_DIM 256

__device__ __forceinline__ unsigned short f2bf(float x) {
    unsigned int u = __float_as_uint(x);
    unsigned int r = u + 0x7FFFu + ((u >> 16) & 1u);   // round-to-nearest-even
    return (unsigned short)(r >> 16);
}

// ---------------- K1: h_proj[b][d] = sum_k hidden[b][k] * attn_w[d][k] + b[d]
__global__ void hproj_kernel(const float* __restrict__ hidden,
                             const float* __restrict__ attn_w,
                             const float* __restrict__ attn_b,
                             float* __restrict__ hproj) {
    int b = blockIdx.x;          // 64 blocks
    int d = threadIdx.x;         // 256 threads
    __shared__ float sh[DEC_DIM];
    sh[d] = hidden[b * DEC_DIM + d];
    __syncthreads();
    const float* wrow = attn_w + (size_t)d * (ENC_DIM + DEC_DIM); // Wh part: cols 0..255
    float acc = attn_b[d];
#pragma unroll 8
    for (int k = 0; k < DEC_DIM; ++k) acc = fmaf(sh[k], wrow[k], acc);
    hproj[b * DEC_DIM + d] = acc;
}

// ---------------- K2: We (attn_w[:,256:768]) -> bf16, row-major [256][512]
__global__ void webf_kernel(const float* __restrict__ attn_w,
                            unsigned short* __restrict__ Webf) {
    int d = blockIdx.x;          // 256 blocks
    int t = threadIdx.x;         // 256 threads
    const float* src = attn_w + (size_t)d * (ENC_DIM + DEC_DIM) + DEC_DIM;
#pragma unroll
    for (int c = t; c < ENC_DIM; c += 256)
        Webf[d * ENC_DIM + c] = f2bf(src[c]);
}

// ---------------- K3: scores[b][n] = tanh(e_proj + h_proj) . v_w  (WMMA core)
// grid (64, 128): block = (batch b, tile of 16 n-rows). 512 threads = 16 waves,
// wave w owns d-columns [16w, 16w+16). K loop over e in steps of 32.
__global__ void __launch_bounds__(512)
scores_kernel(const float* __restrict__ enc,
              const float* __restrict__ hproj,
              const unsigned short* __restrict__ Webf,
              const float* __restrict__ v_w,
              const float* __restrict__ v_b,
              float* __restrict__ scores) {
    const int b  = blockIdx.x;
    const int nt = blockIdx.y;
    const int tid = threadIdx.x;

    __shared__ __align__(32) unsigned short sA[16 * ENC_DIM]; // 16 rows bf16 enc tile
    __shared__ float sScore[16];

    // Stage + convert enc tile (16 x 512 fp32 -> bf16). 16 elems/thread.
    {
        int row = tid >> 5;                // 0..15
        int col = (tid & 31) * 16;         // 0..496
        const float* src = enc + ((size_t)(b * N_DIM + nt * 16 + row)) * ENC_DIM + col;
        unsigned short* dst = &sA[row * ENC_DIM + col];
#pragma unroll
        for (int i = 0; i < 16; i += 4) {
            float4 f = *(const float4*)(src + i);
            dst[i + 0] = f2bf(f.x);
            dst[i + 1] = f2bf(f.y);
            dst[i + 2] = f2bf(f.z);
            dst[i + 3] = f2bf(f.w);
        }
    }
    if (tid < 16) sScore[tid] = 0.0f;
    __syncthreads();

    const int wave = tid >> 5;
    const int lane = tid & 31;
    const int lm   = lane & 15;            // row (A) / column (B) index
    const int hi   = lane >> 4;            // K half selector
    const int dcol = wave * 16 + lm;       // global d for this lane's B column

    const unsigned short* Abase = &sA[lm * ENC_DIM + hi * 16];
    const unsigned short* Bbase = Webf + (size_t)dcol * ENC_DIM + hi * 16;

    v8f acc = {};
#pragma unroll
    for (int k0 = 0; k0 < ENC_DIM; k0 += 32) {
        v16bf a  = *(const v16bf*)(const void*)(Abase + k0);
        v16bf bm = *(const v16bf*)(const void*)(Bbase + k0);
        acc = __builtin_amdgcn_wmma_f32_16x16x32_bf16(
                  false, a, false, bm, (short)0, acc, false, false);
    }

    const float hp  = hproj[b * DEC_DIM + dcol];
    const float vwd = v_w[dcol];

    // D layout: lane l, vgpr v -> (row = v + 8*hi, d = dcol).
    // tanh + weight, then reduce over the 16 d's of this wave (xor 1,2,4,8
    // stays within each 16-lane half), then ds_add_f32 across waves.
#pragma unroll
    for (int v = 0; v < 8; ++v) {
        float t = tanhf(acc[v] + hp) * vwd;
        t += __shfl_xor(t, 1, 32);
        t += __shfl_xor(t, 2, 32);
        t += __shfl_xor(t, 4, 32);
        t += __shfl_xor(t, 8, 32);
        if (lm == 0) atomicAdd(&sScore[v + 8 * hi], t);
    }
    __syncthreads();

    if (tid < 16)
        scores[b * N_DIM + nt * 16 + tid] = sScore[tid] + v_b[0];
}

// ---------------- K4: softmax over n, in-place on the weights region
__global__ void softmax_kernel(float* __restrict__ w) {
    const int b = blockIdx.x;        // 64 blocks
    const int t = threadIdx.x;       // 256 threads, 8 values each
    float* row = w + b * N_DIM;
    __shared__ float sredA[8];
    __shared__ float sredB[8];

    float vals[8];
    float m = -INFINITY;
#pragma unroll
    for (int j = 0; j < 8; ++j) {
        vals[j] = row[t + 256 * j];
        m = fmaxf(m, vals[j]);
    }
#pragma unroll
    for (int s = 16; s >= 1; s >>= 1) m = fmaxf(m, __shfl_xor(m, s, 32));
    if ((t & 31) == 0) sredA[t >> 5] = m;
    __syncthreads();
    if (t == 0) {
        float x = sredA[0];
#pragma unroll
        for (int i = 1; i < 8; ++i) x = fmaxf(x, sredA[i]);
        sredA[0] = x;
    }
    __syncthreads();
    m = sredA[0];

    float s = 0.0f;
#pragma unroll
    for (int j = 0; j < 8; ++j) {
        vals[j] = __expf(vals[j] - m);
        s += vals[j];
    }
#pragma unroll
    for (int sh = 16; sh >= 1; sh >>= 1) s += __shfl_xor(s, sh, 32);
    if ((t & 31) == 0) sredB[t >> 5] = s;
    __syncthreads();
    if (t == 0) {
        float x = 0.0f;
#pragma unroll
        for (int i = 0; i < 8; ++i) x += sredB[i];
        sredB[0] = x;
    }
    __syncthreads();
    const float inv = 1.0f / sredB[0];
#pragma unroll
    for (int j = 0; j < 8; ++j) row[t + 256 * j] = vals[j] * inv;
}

// ---------------- K5: context[b][e] = sum_n weights[b][n] * enc[b][n][e]
// grid (64, 2): 256 threads cover 256 consecutive e (coalesced HBM stream).
__global__ void __launch_bounds__(256)
context_kernel(const float* __restrict__ enc,
               const float* __restrict__ w,
               float* __restrict__ ctx) {
    const int b = blockIdx.x;
    const int e = blockIdx.y * 256 + threadIdx.x;
    __shared__ float sw[N_DIM];
    for (int n = threadIdx.x; n < N_DIM; n += 256) sw[n] = w[b * N_DIM + n];
    __syncthreads();

    const float* base = enc + (size_t)b * N_DIM * ENC_DIM + e;
    float acc = 0.0f;
#pragma unroll 8
    for (int n = 0; n < N_DIM; ++n)
        acc = fmaf(sw[n], base[(size_t)n * ENC_DIM], acc);
    ctx[b * ENC_DIM + e] = acc;
}

// ---------------------------------------------------------------------------
extern "C" void kernel_launch(void* const* d_in, const int* in_sizes, int n_in,
                              void* d_out, int out_size, void* d_ws, size_t ws_size,
                              hipStream_t stream) {
    (void)in_sizes; (void)n_in; (void)out_size; (void)ws_size;
    const float* hidden = (const float*)d_in[0];  // (64,256)
    const float* enc    = (const float*)d_in[1];  // (64,2048,512)
    const float* attn_w = (const float*)d_in[2];  // (256,768)
    const float* attn_b = (const float*)d_in[3];  // (256,)
    const float* v_w    = (const float*)d_in[4];  // (1,256)
    const float* v_b    = (const float*)d_in[5];  // (1,)

    float* ctx     = (float*)d_out;                          // 64*512
    float* weights = (float*)d_out + B_DIM * ENC_DIM;        // 64*2048 (scores->softmax in place)

    float*          hproj = (float*)d_ws;                                  // 64KB
    unsigned short* Webf  = (unsigned short*)((char*)d_ws + 65536);        // 256KB

    hproj_kernel  <<<B_DIM, DEC_DIM, 0, stream>>>(hidden, attn_w, attn_b, hproj);
    webf_kernel   <<<DEC_DIM, 256, 0, stream>>>(attn_w, Webf);
    scores_kernel <<<dim3(B_DIM, N_DIM / 16), 512, 0, stream>>>(enc, hproj, Webf, v_w, v_b, weights);
    softmax_kernel<<<B_DIM, 256, 0, stream>>>(weights);
    context_kernel<<<dim3(B_DIM, 2), 256, 0, stream>>>(enc, weights, ctx);
}
